// RecurrentSheafBlock_65652870087355
// MI455X (gfx1250) — matrix operand order repaired and checked by reading, not compiled, not run on hardware
//
#include <hip/hip_runtime.h>
#include <math.h>

// Problem constants (B=8, L=2048, D=768)
constexpr int kD    = 768;
constexpr int kL    = 2048;
constexpr int kB    = 8;
constexpr int kRows = kB * kL;     // 16384
constexpr int kKT   = kD / 32;     // 24 K-tiles of 32 (bf16 WMMA)
constexpr float kEps = 1e-5f;
constexpr int kScanWGs = 16;       // 16 persistent WGs, 48 columns each

typedef __attribute__((ext_vector_type(16))) __bf16 v16bf;
typedef __attribute__((ext_vector_type(8)))  float  v8f;

union FragB16 { v16bf v; unsigned int u[8]; };

__device__ __forceinline__ unsigned short f2bf(float f) {
  unsigned int u = __float_as_uint(f);
  u += 0x7FFFu + ((u >> 16) & 1u);            // round-to-nearest-even
  return (unsigned short)(u >> 16);
}
__device__ __forceinline__ float bf2f(unsigned short h) {
  return __uint_as_float(((unsigned int)h) << 16);
}
__device__ __forceinline__ unsigned int pack2bf(float lo, float hi) {
  return (unsigned int)f2bf(lo) | ((unsigned int)f2bf(hi) << 16);
}
__device__ __forceinline__ float sigmoidf(float x) { return 1.0f / (1.0f + __expf(-x)); }

// ---- A fragment: 16x32 bf16 tile, row-major source (ushort), leading dim ldk.
// Layout used here (contiguous K, same as B): lanes 0-15 -> M=lane, lane-half h holds K = 16h..16h+15 contiguously.
__device__ __forceinline__ v16bf load_fragA(const unsigned short* a, int ldk, int kbase, int lane) {
  FragB16 f;
  const int m  = lane & 15;
  const int hi = lane >> 4;                   // 0: K0-15, 1: K16-31
  const unsigned short* r = a + m * ldk + kbase + hi * 16;
#pragma unroll
  for (int v = 0; v < 4; ++v) f.u[v]     = *(const unsigned int*)(r + v * 2);
#pragma unroll
  for (int v = 0; v < 4; ++v) f.u[4 + v] = *(const unsigned int*)(r + 8 + v * 2);
  return f.v;
}

// ---- B fragment for  x @ W^T :  B(k,n) = W[n][k],  W row-major bf16 [n][ldk].
// Layout used here (contiguous K, same as A): N = lane&15; lanes 0-15 hold K=0..15, lanes 16-31 hold K=16..31.
__device__ __forceinline__ v16bf load_fragB(const unsigned short* w, int ldk, int nbase, int kbase, int lane) {
  FragB16 f;
  const int n  = lane & 15;
  const int hi = lane >> 4;
  const unsigned short* r = w + (size_t)(nbase + n) * ldk + kbase + hi * 16;
#pragma unroll
  for (int v = 0; v < 8; ++v) f.u[v] = *(const unsigned int*)(r + v * 2);
  return f.v;
}

// ---- Same B fragment but sourced from the fp32 weight matrix (convert on load).
__device__ __forceinline__ v16bf load_fragB_f32(const float* w, int ldk, int nbase, int kbase, int lane) {
  FragB16 f;
  const int n  = lane & 15;
  const int hi = lane >> 4;
  const float* r = w + (size_t)(nbase + n) * ldk + kbase + hi * 16;
#pragma unroll
  for (int v = 0; v < 8; ++v) f.u[v] = pack2bf(r[v * 2], r[v * 2 + 1]);
  return f.v;
}

__device__ __forceinline__ v8f wmma_bf16(v16bf a, v16bf b, v8f c) {
  return __builtin_amdgcn_wmma_f32_16x16x32_bf16(false, a, false, b, (short)0, c, false, false);
}

// ================= fp32 -> bf16 weight conversion =================
__global__ void cvt_kernel(const float* __restrict__ a, unsigned short* __restrict__ b, int n) {
  int i = blockIdx.x * 256 + threadIdx.x;
  if (i < n) b[i] = f2bf(a[i]);
}

// ================= Phase 1: Z = sigmoid(X @ Wg^T + bg)  (bf16 out) ==========
// 256 threads = 8 waves; WG covers a 32-row M-tile (2 A-frags/wave), each wave
// 6 N-tiles (96 cols). B fragments double-buffered: 12 WMMAs per loaded B set.
__global__ __launch_bounds__(256) void gate_kernel(const float* __restrict__ X,
    const unsigned short* __restrict__ Wg, const float* __restrict__ bg,
    unsigned short* __restrict__ Z) {
  __shared__ __align__(16) unsigned short xt[32 * kD];   // 48 KB
  const int tid = threadIdx.x;
  const int rowbase = blockIdx.x * 32;
  const float* xsrc = X + (size_t)rowbase * kD;
  for (int i = tid; i < 32 * kD; i += 256) xt[i] = f2bf(xsrc[i]);
  __syncthreads();

  const int wave = tid >> 5, lane = tid & 31;
  const int nwb = wave * 96;
  v8f acc[2][6];
#pragma unroll
  for (int mt = 0; mt < 2; ++mt)
#pragma unroll
    for (int j = 0; j < 6; ++j)
#pragma unroll
      for (int e = 0; e < 8; ++e) acc[mt][j][e] = 0.f;

  v16bf b0[6], b1[6];
#pragma unroll
  for (int j = 0; j < 6; ++j) b0[j] = load_fragB(Wg, kD, nwb + j * 16, 0, lane);

  for (int kt = 0; kt < kKT; kt += 2) {
    if (kt + 4 < kKT)   // speculative L0 warm-up of an upcoming weight row
      __builtin_prefetch(Wg + (size_t)(nwb + (lane & 15)) * kD + (kt + 4) * 32, 0, 1);
#pragma unroll
    for (int j = 0; j < 6; ++j) b1[j] = load_fragB(Wg, kD, nwb + j * 16, (kt + 1) * 32, lane);
    v16bf a0 = load_fragA(xt, kD, kt * 32, lane);
    v16bf a1 = load_fragA(xt + 16 * kD, kD, kt * 32, lane);
#pragma unroll
    for (int j = 0; j < 6; ++j) {
      acc[0][j] = wmma_bf16(a0, b0[j], acc[0][j]);
      acc[1][j] = wmma_bf16(a1, b0[j], acc[1][j]);
    }
    if (kt + 2 < kKT) {
#pragma unroll
      for (int j = 0; j < 6; ++j) b0[j] = load_fragB(Wg, kD, nwb + j * 16, (kt + 2) * 32, lane);
    }
    a0 = load_fragA(xt, kD, (kt + 1) * 32, lane);
    a1 = load_fragA(xt + 16 * kD, kD, (kt + 1) * 32, lane);
#pragma unroll
    for (int j = 0; j < 6; ++j) {
      acc[0][j] = wmma_bf16(a0, b1[j], acc[0][j]);
      acc[1][j] = wmma_bf16(a1, b1[j], acc[1][j]);
    }
  }

  const int n = lane & 15, mext = (lane >> 4) * 8;
#pragma unroll
  for (int mt = 0; mt < 2; ++mt)
#pragma unroll
    for (int j = 0; j < 6; ++j) {
      const int col = nwb + j * 16 + n;
      const float bgc = bg[col];
#pragma unroll
      for (int v = 0; v < 8; ++v) {
        const size_t row = (size_t)rowbase + mt * 16 + mext + v;
        Z[row * kD + col] = f2bf(sigmoidf(acc[mt][j][v] + bgc));
      }
    }
}

// ================= Phase 2: sequential scan, 16 persistent WGs ==============
// WG i owns output columns [i*48, i*48+48). Each wave keeps its full 16-col x
// 768-K W_r B-tile in REGISTERS (24 frags = 192 VGPRs) for all 2048 steps:
// the serial critical path is just 24 A-frag loads from the L2-resident 24 KB
// ping-pong h buffer + 24 chained WMMAs + elementwise update + device barrier.
__global__ __launch_bounds__(96) void scan_kernel(const float* __restrict__ X,
    const float* __restrict__ Wr, const float* __restrict__ br,
    const float* __restrict__ decay, const unsigned short* __restrict__ Z,
    float* __restrict__ Y, float* __restrict__ Hf, unsigned short* __restrict__ Hb,
    unsigned int* __restrict__ flag) {
  const int tid = threadIdx.x;
  const int ncol0 = blockIdx.x * 48;
  const int wave = tid >> 5, lane = tid & 31;
  const int col = ncol0 + wave * 16 + (lane & 15);
  const float dc  = sigmoidf(decay[col]);
  const float brc = br[col];

  // Register-resident W_r column tile (converted fp32 -> bf16 once).
  v16bf breg[kKT];
#pragma unroll
  for (int kt = 0; kt < kKT; ++kt)
    breg[kt] = load_fragB_f32(Wr, kD, ncol0 + wave * 16, kt * 32, lane);

  for (int t = 0; t < kL; ++t) {
    const int p = t & 1, q = p ^ 1;
    const unsigned short* hsrc = Hb + (size_t)p * 16 * kD;

    v8f acc;
#pragma unroll
    for (int e = 0; e < 8; ++e) acc[e] = 0.f;
#pragma unroll
    for (int kt = 0; kt < kKT; ++kt) {
      v16bf a = load_fragA(hsrc, kD, kt * 32, lane);
      acc = wmma_bf16(a, breg[kt], acc);
    }

    if (lane < 16) {                 // lanes 0-15 hold batch rows 0-7 (v = m)
      const float* hfp = Hf + (size_t)p * 8 * kD;
      float* hfq = Hf + (size_t)q * 8 * kD;
      unsigned short* hbq = Hb + (size_t)q * 16 * kD;
#pragma unroll
      for (int v = 0; v < 8; ++v) {
        const size_t ridx = ((size_t)v * kL + t) * kD + col;
        const float pred = acc[v] + brc;
        const float err  = tanhf(X[ridx] - pred);
        const float zf   = bf2f(Z[ridx]);
        const float hn   = dc * hfp[v * kD + col] + (1.f - dc) * (zf * err);
        hfq[v * kD + col] = hn;
        hbq[v * kD + col] = f2bf(hn);
        Y[ridx] = hn;
      }
    }

    __threadfence();                 // release this thread's h/Y stores
    __syncthreads();                 // whole WG has released
    if (tid == 0) {                  // device-scope join of the 16 WGs
      atomicAdd(flag, 1u);
      const unsigned int goal = (unsigned)kScanWGs * (unsigned)(t + 1);
      volatile unsigned int* vf = flag;
      while (*vf < goal) { __builtin_amdgcn_s_sleep(2); }
    }
    __syncthreads();
    __threadfence();                 // acquire: next step reads peers' h slices
  }
}

// ======== Phase 3: LayerNorm + out = yn @ Wo^T + bo + x  (fused) ============
__global__ __launch_bounds__(256) void lnout_kernel(const float* __restrict__ X,
    const float* __restrict__ Y, const unsigned short* __restrict__ Wo,
    const float* __restrict__ bo, const float* __restrict__ lng,
    const float* __restrict__ lnb, float* __restrict__ Out) {
  extern __shared__ unsigned char smem[];
  float* yf = (float*)smem;                              // [32][768] fp32
  unsigned short* yb = (unsigned short*)(yf + 32 * kD);  // [32][768] bf16 normalized
  __shared__ float mu_s[32], rs_s[32];

  const int tid = threadIdx.x;
  const int rowbase = blockIdx.x * 32;
  const float* ysrc = Y + (size_t)rowbase * kD;
  for (int i = tid; i < 32 * kD; i += 256) yf[i] = ysrc[i];
  __syncthreads();

  const int wave = tid >> 5, lane = tid & 31;
#pragma unroll
  for (int rr = 0; rr < 4; ++rr) {                // each wave: rows w, w+8, w+16, w+24
    const int r = wave + rr * 8;
    float s = 0.f, ss = 0.f;
    for (int i = lane; i < kD; i += 32) { float v = yf[r * kD + i]; s += v; ss += v * v; }
    for (int off = 16; off > 0; off >>= 1) {
      s  += __shfl_down(s, off, 32);
      ss += __shfl_down(ss, off, 32);
    }
    if (lane == 0) {
      const float mu = s / (float)kD;
      const float var = ss / (float)kD - mu * mu;
      mu_s[r] = mu; rs_s[r] = rsqrtf(var + kEps);
    }
  }
  __syncthreads();
  for (int i = tid; i < 32 * kD; i += 256) {
    const int r = i / kD, c = i - r * kD;
    yb[i] = f2bf((yf[i] - mu_s[r]) * rs_s[r] * lng[c] + lnb[c]);
  }
  __syncthreads();

  const int nwb = wave * 96;
  v8f acc[2][6];
#pragma unroll
  for (int mt = 0; mt < 2; ++mt)
#pragma unroll
    for (int j = 0; j < 6; ++j)
#pragma unroll
      for (int e = 0; e < 8; ++e) acc[mt][j][e] = 0.f;

  v16bf b0[6], b1[6];
#pragma unroll
  for (int j = 0; j < 6; ++j) b0[j] = load_fragB(Wo, kD, nwb + j * 16, 0, lane);

  for (int kt = 0; kt < kKT; kt += 2) {
    if (kt + 4 < kKT)
      __builtin_prefetch(Wo + (size_t)(nwb + (lane & 15)) * kD + (kt + 4) * 32, 0, 1);
#pragma unroll
    for (int j = 0; j < 6; ++j) b1[j] = load_fragB(Wo, kD, nwb + j * 16, (kt + 1) * 32, lane);
    v16bf a0 = load_fragA(yb, kD, kt * 32, lane);
    v16bf a1 = load_fragA(yb + 16 * kD, kD, kt * 32, lane);
#pragma unroll
    for (int j = 0; j < 6; ++j) {
      acc[0][j] = wmma_bf16(a0, b0[j], acc[0][j]);
      acc[1][j] = wmma_bf16(a1, b0[j], acc[1][j]);
    }
    if (kt + 2 < kKT) {
#pragma unroll
      for (int j = 0; j < 6; ++j) b0[j] = load_fragB(Wo, kD, nwb + j * 16, (kt + 2) * 32, lane);
    }
    a0 = load_fragA(yb, kD, (kt + 1) * 32, lane);
    a1 = load_fragA(yb + 16 * kD, kD, (kt + 1) * 32, lane);
#pragma unroll
    for (int j = 0; j < 6; ++j) {
      acc[0][j] = wmma_bf16(a0, b1[j], acc[0][j]);
      acc[1][j] = wmma_bf16(a1, b1[j], acc[1][j]);
    }
  }

  const int n = lane & 15, mext = (lane >> 4) * 8;
#pragma unroll
  for (int mt = 0; mt < 2; ++mt)
#pragma unroll
    for (int j = 0; j < 6; ++j) {
      const int col = nwb + j * 16 + n;
      const float boc = bo[col];
#pragma unroll
      for (int v = 0; v < 8; ++v) {
        const size_t row = (size_t)rowbase + mt * 16 + mext + v;
        Out[row * kD + col] = acc[mt][j][v] + boc + X[row * kD + col];
      }
    }
}

extern "C" void kernel_launch(void* const* d_in, const int* in_sizes, int n_in,
                              void* d_out, int out_size, void* d_ws, size_t ws_size,
                              hipStream_t stream) {
  const float* x     = (const float*)d_in[0];
  const float* decay = (const float*)d_in[1];
  const float* W_r   = (const float*)d_in[2];
  const float* b_r   = (const float*)d_in[3];
  const float* W_g   = (const float*)d_in[4];
  const float* b_g   = (const float*)d_in[5];
  const float* W_o   = (const float*)d_in[6];
  const float* b_o   = (const float*)d_in[7];
  const float* ln_g  = (const float*)d_in[8];
  const float* ln_b  = (const float*)d_in[9];
  float* out = (float*)d_out;

  unsigned char* ws = (unsigned char*)d_ws;
  size_t off = 0;
  auto carve = [&](size_t bytes) -> void* {
    void* p = ws + off;
    off += (bytes + 255) & ~(size_t)255;
    return p;
  };
  unsigned short* Wg_bf = (unsigned short*)carve((size_t)kD * kD * 2);
  unsigned short* Wo_bf = (unsigned short*)carve((size_t)kD * kD * 2);
  unsigned short* Z     = (unsigned short*)carve((size_t)kRows * kD * 2);
  float*          Ybuf  = (float*)carve((size_t)kRows * kD * 4);
  float*          Hf    = (float*)carve((size_t)2 * 8 * kD * 4);
  unsigned short* Hb    = (unsigned short*)carve((size_t)2 * 16 * kD * 2);
  unsigned int*   flag  = (unsigned int*)carve(256);

  // zero scan state + barrier counter (stream ops: graph-capture safe)
  hipMemsetAsync(Hf, 0, (size_t)2 * 8 * kD * 4, stream);
  hipMemsetAsync(Hb, 0, (size_t)2 * 16 * kD * 2, stream);
  hipMemsetAsync(flag, 0, 256, stream);

  const int nW = kD * kD;
  cvt_kernel<<<(nW + 255) / 256, 256, 0, stream>>>(W_g, Wg_bf, nW);
  cvt_kernel<<<(nW + 255) / 256, 256, 0, stream>>>(W_o, Wo_bf, nW);

  gate_kernel<<<kRows / 32, 256, 0, stream>>>(x, Wg_bf, b_g, Z);

  scan_kernel<<<kScanWGs, 96, 0, stream>>>(x, W_r, b_r, decay, Z, Ybuf, Hf, Hb, flag);

  const size_t ln_smem = (size_t)32 * kD * 4 + (size_t)32 * kD * 2;  // 144 KB
  lnout_kernel<<<kRows / 32, 256, ln_smem, stream>>>(x, Ybuf, Wo_bf, b_o, ln_g, ln_b, out);
}
